// TransformerBlock_28398323761445
// MI455X (gfx1250) — matrix operand order repaired and checked by reading, compiled, not run on hardware
//
#include <hip/hip_runtime.h>
#include <hip/hip_bf16.h>
#include <math.h>

// ---------------------------------------------------------------------------
// CDNA5 (gfx1250) wave32 WMMA transformer block, v2.
//   GEMM: 2x2 WMMA register tile per wave (4 MACs/K-step), transposed B tile
//         in LDS so fragments load as aligned ds_load_b128.
//   Attention: flash-v2 style; 8 waves share (b,h), K/V tiles staged in LDS.
// ---------------------------------------------------------------------------

typedef __attribute__((ext_vector_type(16))) __bf16          v16bf;
typedef __attribute__((ext_vector_type(8)))  float           v8f;
typedef __attribute__((ext_vector_type(8)))  unsigned short  us8;

union FragBF {
  v16bf          v;
  unsigned short s[16];
  us8            h[2];
};

__device__ __forceinline__ unsigned short f2bf(float f) {
  unsigned int u = __float_as_uint(f);
  u += 0x7FFFu + ((u >> 16) & 1u);        // round-to-nearest-even
  return (unsigned short)(u >> 16);
}

__device__ __forceinline__ v8f bwmma(const FragBF& a, const FragBF& b, v8f c) {
  return __builtin_amdgcn_wmma_f32_16x16x32_bf16(
      false, a.v, false, b.v, (short)0, c, false, false);
}

// ---------------------------------------------------------------------------
// fp32 -> bf16 conversion
// ---------------------------------------------------------------------------
__global__ __launch_bounds__(256) void cvt_bf16_kernel(
    const float* __restrict__ in, unsigned short* __restrict__ out, int n) {
  int i = blockIdx.x * 256 + threadIdx.x;
  int stride = gridDim.x * 256;
  for (; i < n; i += stride) out[i] = f2bf(in[i]);
}

// ---------------------------------------------------------------------------
// LayerNorm: one block per row, bf16 output
// ---------------------------------------------------------------------------
__global__ __launch_bounds__(256) void ln_bf16_kernel(
    const float* __restrict__ x, const float* __restrict__ g,
    const float* __restrict__ b, unsigned short* __restrict__ out, int D) {
  __shared__ float red[256];
  const int row = blockIdx.x;
  const int tid = threadIdx.x;
  const float* xr = x + (long)row * D;

  float s = 0.0f, ss = 0.0f;
  for (int i = tid; i < D; i += 256) { float v = xr[i]; s += v; ss += v * v; }

  red[tid] = s; __syncthreads();
  for (int st = 128; st > 0; st >>= 1) { if (tid < st) red[tid] += red[tid + st]; __syncthreads(); }
  const float mean = red[0] / (float)D;
  __syncthreads();
  red[tid] = ss; __syncthreads();
  for (int st = 128; st > 0; st >>= 1) { if (tid < st) red[tid] += red[tid + st]; __syncthreads(); }
  const float var  = fmaxf(red[0] / (float)D - mean * mean, 0.0f);
  const float rstd = rsqrtf(var + 1e-5f);

  for (int i = tid; i < D; i += 256)
    out[(long)row * D + i] = f2bf((xr[i] - mean) * rstd * g[i] + b[i]);
}

// ---------------------------------------------------------------------------
// bf16 WMMA GEMM: C = A[MxK] @ W[KxN] (+bias, +gelu, +residual)
//   block = 256 threads = 8 waves, block tile 128(M) x 64(N).
//   Waves arranged 4(M) x 2(N); each wave owns a 32x32 register tile
//   -> 4 WMMA per K-step reusing 2 A-frags and 2 B-frags.
//   B tile staged TRANSPOSED in LDS (stride 40 ushorts = 80B, keeps every
//   fragment read a 16B-aligned ds_load_b128).
// EPI bits: 1=bias, 2=gelu(exact), 4=residual(f32), 8=store f32, 16=store bf16
// ---------------------------------------------------------------------------
template <int EPI>
__global__ __launch_bounds__(256) void gemm_bf16_kernel(
    const unsigned short* __restrict__ A, const unsigned short* __restrict__ W,
    const float* __restrict__ bias, const float* __restrict__ res,
    float* __restrict__ outF, unsigned short* __restrict__ outB,
    int M, int N, int K) {
  __shared__ unsigned short BtT[64][40];   // [n][k], padded

  const int tid  = threadIdx.x;
  const int lane = tid & 31;
  const int wid  = tid >> 5;
  const int lo   = lane & 15;
  const int hi   = lane >> 4;

  const int n0 = blockIdx.x * 64;
  const int mW = blockIdx.y * 128 + (wid >> 1) * 32;   // wave row base
  const int nW = (wid & 1) * 32;                       // wave col base in tile

  const int sr = tid >> 3;         // staging: k row 0..31
  const int sc = (tid & 7) * 8;    // staging: n col group

  const long aB0 = (long)(mW + lo) * K;
  const long aB1 = (long)(mW + 16 + lo) * K;

  v8f acc00 = {}, acc01 = {}, acc10 = {}, acc11 = {};

  for (int k0 = 0; k0 < K; k0 += 32) {
    __syncthreads();
    // stage B tile (32 x 64) transposed: one coalesced b128 load per thread
    us8 wrow = *(const us8*)(W + (long)(k0 + sr) * N + (n0 + sc));
#pragma unroll
    for (int i = 0; i < 8; ++i) BtT[sc + i][sr] = wrow[i];
    __syncthreads();

    FragBF a0, a1, b0, b1;
    // A 16x32 bf16 frag: lane<16 -> K {0..15}; lane>=16 -> {16..31}
    a0.h[0] = *(const us8*)(A + aB0 + k0 + hi * 16);
    a0.h[1] = *(const us8*)(A + aB0 + k0 + hi * 16 + 8);
    a1.h[0] = *(const us8*)(A + aB1 + k0 + hi * 16);
    a1.h[1] = *(const us8*)(A + aB1 + k0 + hi * 16 + 8);
    // B 32x16 frag: col = lo, element j -> K = hi*16 + j (contiguous in BtT)
    b0.h[0] = *(const us8*)(&BtT[nW + lo][hi * 16]);
    b0.h[1] = *(const us8*)(&BtT[nW + lo][hi * 16 + 8]);
    b1.h[0] = *(const us8*)(&BtT[nW + 16 + lo][hi * 16]);
    b1.h[1] = *(const us8*)(&BtT[nW + 16 + lo][hi * 16 + 8]);

    acc00 = bwmma(a0, b0, acc00);
    acc01 = bwmma(a0, b1, acc01);
    acc10 = bwmma(a1, b0, acc10);
    acc11 = bwmma(a1, b1, acc11);
  }

  auto epi = [&](v8f acc, int mt, int nt) {
    const int col = nt + lo;
    const float bval = (EPI & 1) ? bias[col] : 0.0f;
#pragma unroll
    for (int r = 0; r < 8; ++r) {
      const long o = (long)(mt + hi * 8 + r) * N + col;   // C row = r + 8*hi
      float val = acc[r] + bval;
      if (EPI & 2)  val = 0.5f * val * (1.0f + erff(val * 0.70710678118654752f));
      if (EPI & 4)  val += res[o];
      if (EPI & 8)  outF[o] = val;
      if (EPI & 16) outB[o] = f2bf(val);
    }
  };
  epi(acc00, mW,      n0 + nW);
  epi(acc01, mW,      n0 + nW + 16);
  epi(acc10, mW + 16, n0 + nW);
  epi(acc11, mW + 16, n0 + nW + 16);
}

// ---------------------------------------------------------------------------
// Flash attention v2: block = 8 waves sharing one (b,h); each wave owns a
// 16-row query tile (block covers 128 query rows). Per iteration the block
// cooperatively stages the shared 32-key K tile and transposed V tile in LDS
// (one coalesced b128 global load per thread each), then:
//   S = Q(16x64) @ K^T  -> 4 WMMA (2 key subtiles x 2 dk-steps)
//   O += P(16x32) @ V   -> 4 WMMA (dk=64 -> 4 column tiles)
// Online softmax per row; P transposed C-frag -> A-frag through LDS.
// ---------------------------------------------------------------------------
__global__ __launch_bounds__(256) void attn_kernel(
    const unsigned short* __restrict__ Q, const unsigned short* __restrict__ Kp,
    const unsigned short* __restrict__ Vp, unsigned short* __restrict__ Ctx,
    int S, int D, int H, float scale) {
  __shared__ unsigned short Kt[32][72];    // [key][dk]   (stride 144B)
  __shared__ unsigned short VtT[64][40];   // [dk][key]   (stride 80B)
  __shared__ float pscr[8][16][33];        // per-wave P tile

  const int tid  = threadIdx.x;
  const int lane = tid & 31;
  const int wid  = tid >> 5;
  const int lo   = lane & 15;
  const int hi   = lane >> 4;

  const int qBlks = S / 128;               // query-blocks per (b,h)
  const int bh    = blockIdx.x / qBlks;
  const int qblk  = blockIdx.x % qBlks;
  const int h = bh % H;
  const int b = bh / H;
  const long base = ((long)b * S) * D + (long)h * 64;
  const int qt = qblk * 8 + wid;           // this wave's 16-row query tile

  // persistent Q fragments (dk 0..31, 32..63)
  FragBF qa0, qa1;
  {
    const unsigned short* qp = Q + base + (long)(qt * 16 + lo) * D;
    qa0.h[0] = *(const us8*)(qp + hi * 16);
    qa0.h[1] = *(const us8*)(qp + hi * 16 + 8);
    qa1.h[0] = *(const us8*)(qp + 32 + hi * 16);
    qa1.h[1] = *(const us8*)(qp + 32 + hi * 16 + 8);
  }

  float mi[8], li[8];
#pragma unroll
  for (int r = 0; r < 8; ++r) { mi[r] = -3.0e38f; li[r] = 0.0f; }
  v8f oacc[4] = {};

  // staging coords: key = tid/8 (0..31), dk group = (tid%8)*8
  const int sk = tid >> 3;
  const int sc = (tid & 7) * 8;
  const unsigned short* kStage = Kp + base + (long)sk * D + sc;
  const unsigned short* vStage = Vp + base + (long)sk * D + sc;
  const long stageStep = (long)32 * D;

  for (int kt = 0; kt < S; kt += 32) {
    __syncthreads();                       // protect LDS tiles (WAR)
    us8 krow = *(const us8*)kStage;
    us8 vrow = *(const us8*)vStage;
    kStage += stageStep;
    vStage += stageStep;
    *(us8*)(&Kt[sk][sc]) = krow;           // K tile row-major
#pragma unroll
    for (int i = 0; i < 8; ++i) VtT[sc + i][sk] = vrow[i];   // V transposed
    __syncthreads();

    // K^T B-frags: col = key, element j -> dk = kk + hi*16 + j (contiguous)
    FragBF bk00, bk01, bk10, bk11;
    bk00.h[0] = *(const us8*)(&Kt[lo][hi * 16]);
    bk00.h[1] = *(const us8*)(&Kt[lo][hi * 16 + 8]);
    bk01.h[0] = *(const us8*)(&Kt[lo][32 + hi * 16]);
    bk01.h[1] = *(const us8*)(&Kt[lo][32 + hi * 16 + 8]);
    bk10.h[0] = *(const us8*)(&Kt[16 + lo][hi * 16]);
    bk10.h[1] = *(const us8*)(&Kt[16 + lo][hi * 16 + 8]);
    bk11.h[0] = *(const us8*)(&Kt[16 + lo][32 + hi * 16]);
    bk11.h[1] = *(const us8*)(&Kt[16 + lo][32 + hi * 16 + 8]);

    v8f s0 = {}, s1 = {};
    s0 = bwmma(qa0, bk00, s0); s0 = bwmma(qa1, bk01, s0);
    s1 = bwmma(qa0, bk10, s1); s1 = bwmma(qa1, bk11, s1);

    // online softmax over the 32 new keys (rows live in 16-lane halves)
    float alpha[8], p0[8], p1[8];
#pragma unroll
    for (int r = 0; r < 8; ++r) {
      float x0 = s0[r] * scale;
      float x1 = s1[r] * scale;
      float mx = fmaxf(x0, x1);
      mx = fmaxf(mx, __shfl_xor(mx, 1, 32));
      mx = fmaxf(mx, __shfl_xor(mx, 2, 32));
      mx = fmaxf(mx, __shfl_xor(mx, 4, 32));
      mx = fmaxf(mx, __shfl_xor(mx, 8, 32));
      const float mn = fmaxf(mi[r], mx);
      const float al = __expf(mi[r] - mn);
      const float e0 = __expf(x0 - mn);
      const float e1 = __expf(x1 - mn);
      float rs = e0 + e1;
      rs += __shfl_xor(rs, 1, 32);
      rs += __shfl_xor(rs, 2, 32);
      rs += __shfl_xor(rs, 4, 32);
      rs += __shfl_xor(rs, 8, 32);
      li[r] = li[r] * al + rs;
      mi[r] = mn;
      alpha[r] = al; p0[r] = e0; p1[r] = e1;
    }

    // rescale accumulators; stage P (C-frag layout) into this wave's LDS slab
#pragma unroll
    for (int r = 0; r < 8; ++r) {
      oacc[0][r] *= alpha[r]; oacc[1][r] *= alpha[r];
      oacc[2][r] *= alpha[r]; oacc[3][r] *= alpha[r];
      pscr[wid][hi * 8 + r][lo]      = p0[r];
      pscr[wid][hi * 8 + r][16 + lo] = p1[r];
    }
    __syncthreads();                       // uniform trip count on all waves

    // read back P as A-fragment (row = lo, K = key within 32-tile)
    FragBF pa;
#pragma unroll
    for (int j = 0; j < 8; ++j) {
      pa.s[j]     = f2bf(pscr[wid][lo][hi * 16 + j]);
      pa.s[8 + j] = f2bf(pscr[wid][lo][hi * 16 + 8 + j]);
    }

    // O += P @ V : V B-frag col = dk (t*16+lo), element j -> key = hi*16 + j
#pragma unroll
    for (int t = 0; t < 4; ++t) {
      FragBF bv;
      bv.h[0] = *(const us8*)(&VtT[t * 16 + lo][hi * 16]);
      bv.h[1] = *(const us8*)(&VtT[t * 16 + lo][hi * 16 + 8]);
      oacc[t] = bwmma(pa, bv, oacc[t]);
    }
  }

  // normalize and write ctx (bf16, [B,S,D] layout)
#pragma unroll
  for (int t = 0; t < 4; ++t) {
#pragma unroll
    for (int r = 0; r < 8; ++r) {
      const float val = oacc[t][r] / li[r];
      Ctx[base + (long)(qt * 16 + hi * 8 + r) * D + t * 16 + lo] = f2bf(val);
    }
  }
}

// ---------------------------------------------------------------------------
// Launch
// ---------------------------------------------------------------------------
extern "C" void kernel_launch(void* const* d_in, const int* in_sizes, int n_in,
                              void* d_out, int out_size, void* d_ws, size_t ws_size,
                              hipStream_t stream) {
  (void)in_sizes; (void)n_in; (void)out_size; (void)ws_size;
  const int Bn = 2, S = 2048, D = 1024, F = 4096, H = 16;
  const int M = Bn * S;  // 4096

  const float* x   = (const float*)d_in[0];
  const float* Wq  = (const float*)d_in[1];
  const float* bq  = (const float*)d_in[2];
  const float* Wk  = (const float*)d_in[3];
  const float* bk  = (const float*)d_in[4];
  const float* Wv  = (const float*)d_in[5];
  const float* bv  = (const float*)d_in[6];
  const float* Wo  = (const float*)d_in[7];
  const float* bo  = (const float*)d_in[8];
  const float* g1  = (const float*)d_in[9];
  const float* be1 = (const float*)d_in[10];
  const float* g2  = (const float*)d_in[11];
  const float* be2 = (const float*)d_in[12];
  const float* W1  = (const float*)d_in[13];
  const float* b1  = (const float*)d_in[14];
  const float* W2  = (const float*)d_in[15];
  const float* b2  = (const float*)d_in[16];
  float* out = (float*)d_out;

  char* wsp = (char*)d_ws;
  size_t off = 0;
  auto alloc = [&](size_t bytes) -> void* {
    void* p = wsp + off;
    off += (bytes + 255) & ~(size_t)255;
    return p;
  };
  unsigned short* hbuf = (unsigned short*)alloc((size_t)M * D * 2);  // LN out
  unsigned short* wqb  = (unsigned short*)alloc((size_t)D * D * 2);
  unsigned short* wkb  = (unsigned short*)alloc((size_t)D * D * 2);
  unsigned short* wvb  = (unsigned short*)alloc((size_t)D * D * 2);
  unsigned short* wob  = (unsigned short*)alloc((size_t)D * D * 2);
  unsigned short* w1b  = (unsigned short*)alloc((size_t)D * F * 2);
  unsigned short* w2b  = (unsigned short*)alloc((size_t)F * D * 2);
  unsigned short* qb   = (unsigned short*)alloc((size_t)M * D * 2);  // contiguous
  unsigned short* kb   = (unsigned short*)alloc((size_t)M * D * 2);  //   block
  unsigned short* vb   = (unsigned short*)alloc((size_t)M * D * 2);
  unsigned short* cb   = (unsigned short*)alloc((size_t)M * D * 2);
  float*          x1   = (float*)alloc((size_t)M * D * 4);
  unsigned short* h3   = qb;   // FFN hidden (M*F) aliases dead q/k/v/ctx region

  // weight fp32 -> bf16
  cvt_bf16_kernel<<<(D * D + 255) / 256, 256, 0, stream>>>(Wq, wqb, D * D);
  cvt_bf16_kernel<<<(D * D + 255) / 256, 256, 0, stream>>>(Wk, wkb, D * D);
  cvt_bf16_kernel<<<(D * D + 255) / 256, 256, 0, stream>>>(Wv, wvb, D * D);
  cvt_bf16_kernel<<<(D * D + 255) / 256, 256, 0, stream>>>(Wo, wob, D * D);
  cvt_bf16_kernel<<<(D * F + 255) / 256, 256, 0, stream>>>(W1, w1b, D * F);
  cvt_bf16_kernel<<<(F * D + 255) / 256, 256, 0, stream>>>(W2, w2b, F * D);

  const dim3 gD(D / 64, M / 128);   // N=1024 GEMMs
  const dim3 gF(F / 64, M / 128);   // N=4096 GEMM

  // --- attention sublayer ---
  ln_bf16_kernel<<<M, 256, 0, stream>>>(x, g1, be1, hbuf, D);
  gemm_bf16_kernel<17><<<gD, 256, 0, stream>>>(hbuf, wqb, bq, nullptr, nullptr, qb, M, D, D);
  gemm_bf16_kernel<17><<<gD, 256, 0, stream>>>(hbuf, wkb, bk, nullptr, nullptr, kb, M, D, D);
  gemm_bf16_kernel<17><<<gD, 256, 0, stream>>>(hbuf, wvb, bv, nullptr, nullptr, vb, M, D, D);
  attn_kernel<<<Bn * H * (S / 128), 256, 0, stream>>>(qb, kb, vb, cb, S, D, H, 0.125f);
  gemm_bf16_kernel<13><<<gD, 256, 0, stream>>>(cb, wob, bo, x, x1, nullptr, M, D, D);

  // --- FFN sublayer ---
  ln_bf16_kernel<<<M, 256, 0, stream>>>(x1, g2, be2, hbuf, D);
  gemm_bf16_kernel<19><<<gF, 256, 0, stream>>>(hbuf, w1b, b1, nullptr, nullptr, h3, M, F, D);
  gemm_bf16_kernel<13><<<gD, 256, 0, stream>>>(h3, w2b, b2, x1, out, nullptr, M, D, F);
}
